// IG_MSA_11690900980038
// MI455X (gfx1250) — matrix operand order repaired and checked by reading, not compiled, not run on hardware
//
#include <hip/hip_runtime.h>
#include <hip/hip_bf16.h>

// IG-MSA (Retinexformer) for MI455X / gfx1250.
// All four big GEMMs run on v_wmma_f32_16x16x32_f16 with fp32 accumulation.
// b=4, H=W=128 -> M=65536 tokens, c=512, heads=8, dh=64.
// Fragments load as ds_load_b128 pairs; K-tile = 64 (one barrier pair per
// 12 WMMAs); all B fragments hoisted so WMMAs issue back-to-back.

typedef __attribute__((ext_vector_type(16))) _Float16 v16h;
typedef __attribute__((ext_vector_type(8)))  float    v8f;

union F16x16 { v16h v; uint4 q[2]; };
union F16x8  { uint4 q; _Float16 h[8]; };

// A matrix 16x32 (MxK), row-major src[m*ld + k].
// Lane: m=lane&15, ko=(lane>>4)*8.  Elements 0..7 = K ko..ko+7, elements
// 8..15 = K ko+16..ko+23 -> two b128 LDS loads.
__device__ __forceinline__ v16h frag_a(const _Float16* src, int ld) {
  int lane = threadIdx.x & 31;
  int m = lane & 15;
  int ko = (lane >> 4) * 8;
  F16x16 u;
  u.q[0] = *(const uint4*)&src[m * ld + ko];
  u.q[1] = *(const uint4*)&src[m * ld + ko + 16];
  return u.v;
}

// B matrix 32x16 (KxN) with the LDS tile stored TRANSPOSED:
// src[n*ld + k] == B[k][n].  Lane: n=lane&15, ko=(lane>>4)*8.
// Elements 0..7 = K ko..ko+7, elements 8..15 = K ko+16..ko+23 -> two b128 LDS loads.
__device__ __forceinline__ v16h frag_bT(const _Float16* src, int ld) {
  int lane = threadIdx.x & 31;
  int n = lane & 15;
  int ko = (lane >> 4) * 8;
  F16x16 u;
  u.q[0] = *(const uint4*)&src[n * ld + ko];
  u.q[1] = *(const uint4*)&src[n * ld + ko + 16];
  return u.v;
}

// C/D layout: VGPR i, lanes0-15 -> M=i, lanes16-31 -> M=8+i; N = lane&15.

// ---------------- small utility kernels ----------------
__global__ __launch_bounds__(256) void zero_f32(float* p, int n) {
  int i = blockIdx.x * 256 + threadIdx.x;
  if (i < n) p[i] = 0.f;
}

__global__ __launch_bounds__(256) void cvt_f32_f16(const float* __restrict__ in,
                                                   _Float16* __restrict__ out, int n) {
  int i = blockIdx.x * 256 + threadIdx.x;
  if (i < n) out[i] = (_Float16)in[i];
}

// Convert 512x512 f32 weight (row-major [k][n]) to f16 stored transposed [n][k].
__global__ __launch_bounds__(256) void cvt_transpose(const float* __restrict__ in,
                                                     _Float16* __restrict__ out) {
  int i = blockIdx.x * 256 + threadIdx.x;  // 262144 total
  int k = i >> 9, n = i & 511;
  out[(size_t)n * 512 + k] = (_Float16)in[i];
}

// ---------------- fused QKV projection GEMM ----------------
// X[65536x512]f16 @ {Wq,Wk,Wv} (f16, pre-transposed [n][k]).
// Block tile 64x64, K-tile 64, 8 waves: 2 subtiles x 3 matrices each.
// Epilogue: Q,K stored CHANNEL-MAJOR [b][c][n] f16 (one b128 store per lane)
// + sum-of-squares atomics; V gated by illu -> f16 token-major; raw V -> f32.
__global__ __launch_bounds__(256)
void qkv_gemm(const _Float16* __restrict__ Xh,
              const _Float16* __restrict__ Wtq, const _Float16* __restrict__ Wtk,
              const _Float16* __restrict__ Wtv, const float* __restrict__ illu,
              _Float16* __restrict__ Qt, _Float16* __restrict__ Kt,
              _Float16* __restrict__ Vg, float* __restrict__ Vinp,
              float* __restrict__ qss, float* __restrict__ kss) {
  const int LDA = 72, LDB = 72;
  __shared__ _Float16 sA[64 * LDA];        // A tile [64 rows][64 k]
  __shared__ _Float16 sBt[3][64 * LDB];    // B tiles transposed [64 n][64 k]
  const int bm = blockIdx.x * 64;
  const int bn = blockIdx.y * 64;
  const int t = threadIdx.x;
  const int w = t >> 5, rg = w & 3, cg = w >> 2;
  v8f acc[3][2] = {};

  const int r = t >> 2, seg = (t & 3) * 16;
  for (int kk = 0; kk < 512; kk += 64) {
    {
      const uint4* ga = (const uint4*)&Xh[(size_t)(bm + r) * 512 + kk + seg];
      uint4* la = (uint4*)&sA[r * LDA + seg];
      la[0] = ga[0]; la[1] = ga[1];
      const _Float16* Ws[3] = {Wtq, Wtk, Wtv};
#pragma unroll
      for (int i = 0; i < 3; ++i) {
        const uint4* gb = (const uint4*)&Ws[i][(size_t)(bn + r) * 512 + kk + seg];
        uint4* lb = (uint4*)&sBt[i][r * LDB + seg];
        lb[0] = gb[0]; lb[1] = gb[1];
      }
      if (kk + 64 < 512)
        __builtin_prefetch(&Xh[(size_t)(bm + r) * 512 + kk + 64 + seg], 0, 1);
    }
    __syncthreads();
#pragma unroll
    for (int ks = 0; ks < 64; ks += 32) {
      v16h a = frag_a(&sA[(16 * rg) * LDA + ks], LDA);
      v16h bf[6];
#pragma unroll
      for (int j = 0; j < 2; ++j)
#pragma unroll
        for (int i = 0; i < 3; ++i)
          bf[j * 3 + i] = frag_bT(&sBt[i][(32 * cg + 16 * j) * LDB + ks], LDB);
#pragma unroll
      for (int j = 0; j < 2; ++j)
#pragma unroll
        for (int i = 0; i < 3; ++i)
          acc[i][j] = __builtin_amdgcn_wmma_f32_16x16x32_f16(
              false, a, false, bf[j * 3 + i], (short)0, acc[i][j], false, false);
    }
    __syncthreads();
  }

  const int lane = t & 31, nn = lane & 15, mb = (lane >> 4) * 8;
  const int bidx = bm >> 14;               // 16384 tokens per image
  const int rowloc0 = (bm & 16383) + 16 * rg + mb;
#pragma unroll
  for (int j = 0; j < 2; ++j) {
    int col = bn + 32 * cg + 16 * j + nn;
    int rowb = bm + 16 * rg + mb;
    F16x8 qv, kv;
    float ssq = 0.f, ssk = 0.f;
#pragma unroll
    for (int i = 0; i < 8; ++i) {
      float q = acc[0][j][i], k = acc[1][j][i], vr = acc[2][j][i];
      qv.h[i] = (_Float16)q;
      kv.h[i] = (_Float16)k;
      ssq += q * q;
      ssk += k * k;
      size_t idx = (size_t)(rowb + i) * 512 + col;
      Vinp[idx] = vr;
      Vg[idx] = (_Float16)(vr * illu[idx]);
    }
    size_t tbase = ((size_t)(bidx * 512 + col)) * 16384 + rowloc0;
    *(uint4*)&Qt[tbase] = qv.q;            // channel-major, b128 store
    *(uint4*)&Kt[tbase] = kv.q;
    ssq += __shfl_xor(ssq, 16, 32);
    ssk += __shfl_xor(ssk, 16, 32);
    if (lane < 16) {
      atomicAdd(&qss[bidx * 512 + col], ssq);
      atomicAdd(&kss[bidx * 512 + col], ssk);
    }
  }
}

// ---------------- Gram: G[b,h,d,e] = sum_n K[b,c=h*64+d,n]*Q[b,c=h*64+e,n] ----
// Q/K are channel-major so both LDS tiles are [channel][token]; K-tile 64.
// grid (32 bh, 16 k-splits of 1024 tokens); fp32 atomic accumulate.
__global__ __launch_bounds__(256)
void gram_kernel(const _Float16* __restrict__ Kt, const _Float16* __restrict__ Qt,
                 float* __restrict__ G) {
  const int LD = 72;
  __shared__ _Float16 sK[64 * LD];         // [d][n]
  __shared__ _Float16 sQ[64 * LD];         // [e][n]
  const int bh = blockIdx.x, b = bh >> 3, h = bh & 7;
  const int nbase = blockIdx.y * 1024;
  const size_t cbase = (size_t)(b * 512 + h * 64) * 16384;
  const int t = threadIdx.x;
  const int w = t >> 5, rg = w & 3, cg = w >> 2;
  v8f acc[2] = {};

  const int c = t >> 2, seg = (t & 3) * 16;
  for (int n0 = 0; n0 < 1024; n0 += 64) {
    size_t gidx = cbase + (size_t)c * 16384 + nbase + n0 + seg;
    const uint4* gk = (const uint4*)&Kt[gidx];
    uint4* lk = (uint4*)&sK[c * LD + seg];
    lk[0] = gk[0]; lk[1] = gk[1];
    const uint4* gq = (const uint4*)&Qt[gidx];
    uint4* lq = (uint4*)&sQ[c * LD + seg];
    lq[0] = gq[0]; lq[1] = gq[1];
    if (n0 + 64 < 1024) __builtin_prefetch(&Kt[gidx + 64], 0, 1);
    __syncthreads();
#pragma unroll
    for (int ks = 0; ks < 64; ks += 32) {
      v16h a = frag_a(&sK[(16 * rg) * LD + ks], LD);   // A[d][n]
      v16h bf[2];
#pragma unroll
      for (int j = 0; j < 2; ++j)
        bf[j] = frag_bT(&sQ[(32 * cg + 16 * j) * LD + ks], LD);  // B[n][e]
#pragma unroll
      for (int j = 0; j < 2; ++j)
        acc[j] = __builtin_amdgcn_wmma_f32_16x16x32_f16(
            false, a, false, bf[j], (short)0, acc[j], false, false);
    }
    __syncthreads();
  }

  const int lane = t & 31, nn = lane & 15, mb = (lane >> 4) * 8;
#pragma unroll
  for (int j = 0; j < 2; ++j)
#pragma unroll
    for (int i = 0; i < 8; ++i)
      atomicAdd(&G[(size_t)bh * 4096 + (16 * rg + mb + i) * 64 + 32 * cg + 16 * j + nn],
                acc[j][i]);
}

// ---------------- scale by 1/(||k||*||q||), rescale[h], softmax over e --------
__global__ __launch_bounds__(64)
void softmax_kernel(const float* __restrict__ G, const float* __restrict__ qss,
                    const float* __restrict__ kss, const float* __restrict__ rescale,
                    _Float16* __restrict__ Ah) {
  const int bh = blockIdx.x, b = bh >> 3, h = bh & 7;
  const int d = threadIdx.x;
  const float eps = 1e-12f;
  const float kn = fmaxf(sqrtf(kss[b * 512 + h * 64 + d]), eps);
  const float rs = rescale[h];
  float vals[64];
  float mx = -1e30f;
  for (int e = 0; e < 64; ++e) {
    float qn = fmaxf(sqrtf(qss[b * 512 + h * 64 + e]), eps);
    float v = G[(size_t)bh * 4096 + d * 64 + e] * rs / (kn * qn);
    vals[e] = v;
    mx = fmaxf(mx, v);
  }
  float s = 0.f;
  for (int e = 0; e < 64; ++e) { vals[e] = expf(vals[e] - mx); s += vals[e]; }
  float inv = 1.f / s;
  for (int e = 0; e < 64; ++e)
    Ah[(size_t)bh * 4096 + d * 64 + e] = (_Float16)(vals[e] * inv);
}

// ---------------- xo[b,n,h*64+d] = sum_e attn[b,h,d,e] * Vg[b,n,h*64+e] -------
__global__ __launch_bounds__(256)
void attnv_kernel(const _Float16* __restrict__ Vg, const _Float16* __restrict__ Ah,
                  _Float16* __restrict__ Xo) {
  const int LDV = 72, LDT = 72;
  __shared__ _Float16 sV[64 * LDV];        // [token][e]
  __shared__ _Float16 sAt[64 * LDT];       // [d][e]
  const int rowb = blockIdx.x * 64;
  const int h = blockIdx.y;
  const int b = rowb >> 14;
  const int t = threadIdx.x;
  const int w = t >> 5, rg = w & 3, cg = w >> 2;
  {
    int r = t >> 2, seg = (t & 3) * 16;
    const uint4* gv = (const uint4*)&Vg[(size_t)(rowb + r) * 512 + h * 64 + seg];
    uint4* sv = (uint4*)&sV[r * LDV + seg];
    sv[0] = gv[0]; sv[1] = gv[1];
    const uint4* ga = (const uint4*)&Ah[(size_t)(b * 8 + h) * 4096 + r * 64 + seg];
    uint4* sa = (uint4*)&sAt[r * LDT + seg];
    sa[0] = ga[0]; sa[1] = ga[1];
  }
  __syncthreads();
  v8f acc[2] = {};
#pragma unroll
  for (int ke = 0; ke < 64; ke += 32) {
    v16h a = frag_a(&sV[(16 * rg) * LDV + ke], LDV);
    v16h bf[2];
#pragma unroll
    for (int j = 0; j < 2; ++j)
      bf[j] = frag_bT(&sAt[(32 * cg + 16 * j) * LDT + ke], LDT);  // B[e][d]
#pragma unroll
    for (int j = 0; j < 2; ++j)
      acc[j] = __builtin_amdgcn_wmma_f32_16x16x32_f16(
          false, a, false, bf[j], (short)0, acc[j], false, false);
  }
  const int lane = t & 31, nn = lane & 15, mb = (lane >> 4) * 8;
#pragma unroll
  for (int j = 0; j < 2; ++j)
#pragma unroll
    for (int i = 0; i < 8; ++i)
      Xo[(size_t)(rowb + 16 * rg + mb + i) * 512 + h * 64 + 32 * cg + 16 * j + nn] =
          (_Float16)acc[j][i];
}

// ---------------- output projection: out = Xo @ Wp + bp (fp32 out) ------------
__global__ __launch_bounds__(256)
void proj_gemm(const _Float16* __restrict__ Xo, const _Float16* __restrict__ Wtp,
               const float* __restrict__ bp, float* __restrict__ out) {
  const int LDA = 72, LDB = 72;
  __shared__ _Float16 sA[64 * LDA];
  __shared__ _Float16 sBt[64 * LDB];       // transposed [n][k]
  const int bm = blockIdx.x * 64;
  const int bn = blockIdx.y * 64;
  const int t = threadIdx.x;
  const int w = t >> 5, rg = w & 3, cg = w >> 2;
  v8f acc[2] = {};

  const int r = t >> 2, seg = (t & 3) * 16;
  for (int kk = 0; kk < 512; kk += 64) {
    {
      const uint4* ga = (const uint4*)&Xo[(size_t)(bm + r) * 512 + kk + seg];
      uint4* la = (uint4*)&sA[r * LDA + seg];
      la[0] = ga[0]; la[1] = ga[1];
      const uint4* gb = (const uint4*)&Wtp[(size_t)(bn + r) * 512 + kk + seg];
      uint4* lb = (uint4*)&sBt[r * LDB + seg];
      lb[0] = gb[0]; lb[1] = gb[1];
      if (kk + 64 < 512)
        __builtin_prefetch(&Xo[(size_t)(bm + r) * 512 + kk + 64 + seg], 0, 1);
    }
    __syncthreads();
#pragma unroll
    for (int ks = 0; ks < 64; ks += 32) {
      v16h a = frag_a(&sA[(16 * rg) * LDA + ks], LDA);
      v16h bf[2];
#pragma unroll
      for (int j = 0; j < 2; ++j)
        bf[j] = frag_bT(&sBt[(32 * cg + 16 * j) * LDB + ks], LDB);
#pragma unroll
      for (int j = 0; j < 2; ++j)
        acc[j] = __builtin_amdgcn_wmma_f32_16x16x32_f16(
            false, a, false, bf[j], (short)0, acc[j], false, false);
    }
    __syncthreads();
  }

  const int lane = t & 31, nn = lane & 15, mb = (lane >> 4) * 8;
#pragma unroll
  for (int j = 0; j < 2; ++j) {
    int col = bn + 32 * cg + 16 * j + nn;
    float bias = bp[col];
#pragma unroll
    for (int i = 0; i < 8; ++i)
      out[(size_t)(bm + 16 * rg + mb + i) * 512 + col] = acc[j][i] + bias;
  }
}

// ---------------- depthwise 3x3 convs, NHWC, SAME padding ---------------------
__global__ __launch_bounds__(256)
void dwconv_gelu(const float* __restrict__ in, const float* __restrict__ wgt,
                 float* __restrict__ out) {
  size_t i = (size_t)blockIdx.x * 256 + threadIdx.x;
  int ch = (int)(i & 511);
  int x = (int)((i >> 9) & 127);
  int y = (int)((i >> 16) & 127);
  size_t bb = i >> 23;
  const float* wp = &wgt[ch * 9];
  float s = 0.f;
#pragma unroll
  for (int ky = 0; ky < 3; ++ky) {
    int yy = y + ky - 1;
    if (yy < 0 || yy > 127) continue;
#pragma unroll
    for (int kx = 0; kx < 3; ++kx) {
      int xx = x + kx - 1;
      if (xx < 0 || xx > 127) continue;
      s += in[(bb << 23) | ((size_t)yy << 16) | ((size_t)xx << 9) | ch] * wp[ky * 3 + kx];
    }
  }
  out[i] = 0.5f * s * (1.f + erff(s * 0.70710678118f));  // exact GELU
}

__global__ __launch_bounds__(256)
void dwconv_add(const float* __restrict__ in, const float* __restrict__ wgt,
                float* __restrict__ out) {
  size_t i = (size_t)blockIdx.x * 256 + threadIdx.x;
  int ch = (int)(i & 511);
  int x = (int)((i >> 9) & 127);
  int y = (int)((i >> 16) & 127);
  size_t bb = i >> 23;
  const float* wp = &wgt[ch * 9];
  float s = 0.f;
#pragma unroll
  for (int ky = 0; ky < 3; ++ky) {
    int yy = y + ky - 1;
    if (yy < 0 || yy > 127) continue;
#pragma unroll
    for (int kx = 0; kx < 3; ++kx) {
      int xx = x + kx - 1;
      if (xx < 0 || xx > 127) continue;
      s += in[(bb << 23) | ((size_t)yy << 16) | ((size_t)xx << 9) | ch] * wp[ky * 3 + kx];
    }
  }
  out[i] += s;  // stream-ordered after proj_gemm wrote out_c
}

// ---------------- launch ----------------
extern "C" void kernel_launch(void* const* d_in, const int* in_sizes, int n_in,
                              void* d_out, int out_size, void* d_ws, size_t ws_size,
                              hipStream_t stream) {
  const float* x    = (const float*)d_in[0];
  const float* illu = (const float*)d_in[1];
  const float* Wq   = (const float*)d_in[2];
  const float* Wk   = (const float*)d_in[3];
  const float* Wv   = (const float*)d_in[4];
  const float* resc = (const float*)d_in[5];
  const float* Wp   = (const float*)d_in[6];
  const float* bp   = (const float*)d_in[7];
  const float* w1   = (const float*)d_in[8];
  const float* w2   = (const float*)d_in[9];
  float* out = (float*)d_out;

  const size_t M = 65536, C = 512;
  size_t off = 0;
  auto carve = [&](size_t bytes) -> char* {
    char* p = (char*)d_ws + off;
    off += (bytes + 255) & ~(size_t)255;
    return p;
  };
  _Float16* Xh  = (_Float16*)carve(M * C * 2);
  _Float16* Qt  = (_Float16*)carve(M * C * 2);   // channel-major [b][c][n]
  _Float16* Kt  = (_Float16*)carve(M * C * 2);   // channel-major [b][c][n]
  _Float16* Vg  = (_Float16*)carve(M * C * 2);   // token-major
  _Float16* Xo  = (_Float16*)carve(M * C * 2);   // token-major
  _Float16* Wtq = (_Float16*)carve(C * C * 2);   // transposed [n][k]
  _Float16* Wtk = (_Float16*)carve(C * C * 2);
  _Float16* Wtv = (_Float16*)carve(C * C * 2);
  _Float16* Wtp = (_Float16*)carve(C * C * 2);
  float* Vinp = (float*)carve(M * C * 4);
  float* T    = (float*)carve(M * C * 4);
  float* qss  = (float*)carve(2048 * 4);         // contiguous with kss, G
  float* kss  = (float*)carve(2048 * 4);
  float* G    = (float*)carve(32 * 64 * 64 * 4);
  _Float16* Ah = (_Float16*)carve(32 * 64 * 64 * 2);

  // zero qss + kss + G in one shot (carved contiguously)
  zero_f32<<<(135168 + 255) / 256, 256, 0, stream>>>(qss, 135168);

  cvt_f32_f16<<<(int)((M * C + 255) / 256), 256, 0, stream>>>(x, Xh, (int)(M * C));
  cvt_transpose<<<(int)((C * C + 255) / 256), 256, 0, stream>>>(Wq, Wtq);
  cvt_transpose<<<(int)((C * C + 255) / 256), 256, 0, stream>>>(Wk, Wtk);
  cvt_transpose<<<(int)((C * C + 255) / 256), 256, 0, stream>>>(Wv, Wtv);
  cvt_transpose<<<(int)((C * C + 255) / 256), 256, 0, stream>>>(Wp, Wtp);

  qkv_gemm<<<dim3(1024, 8), 256, 0, stream>>>(Xh, Wtq, Wtk, Wtv, illu,
                                              Qt, Kt, Vg, Vinp, qss, kss);
  gram_kernel<<<dim3(32, 16), 256, 0, stream>>>(Kt, Qt, G);
  softmax_kernel<<<32, 64, 0, stream>>>(G, qss, kss, resc, Ah);
  attnv_kernel<<<dim3(1024, 8), 256, 0, stream>>>(Vg, Ah, Xo);
  proj_gemm<<<dim3(1024, 8), 256, 0, stream>>>(Xo, Wtp, bp, out);

  dwconv_gelu<<<131072, 256, 0, stream>>>(Vinp, w1, T);
  dwconv_add<<<131072, 256, 0, stream>>>(T, w2, out);
}
